// MoleculeAtt_84885733638664
// MI455X (gfx1250) — matrix operand order repaired and checked by reading, not compiled, not run on hardware
//
#include <hip/hip_runtime.h>
#include <hip/hip_bf16.h>
#include <stdint.h>

// ---------------------------------------------------------------------------
// MoleculeAtt for MI455X (gfx1250): bf16 WMMA GEMMs + async-to-LDS staging
// + L2-resident bf16 attention.
// ---------------------------------------------------------------------------

typedef __bf16 bf16_t;
typedef __attribute__((ext_vector_type(16))) __bf16 v16bf;
typedef __attribute__((ext_vector_type(8)))  __bf16 bf16x8;
typedef __attribute__((ext_vector_type(8)))  float  v8f;
typedef __attribute__((ext_vector_type(4)))  float  f32x4;

#define B_  512
#define T_  256
#define F_  256
#define E_  256
#define H_  256
#define KQ  768    // q_star width: [q(256) | r(256) | x(256)]
#define NG  1024   // 4*H

// Async copy: 16 bytes/lane, global -> LDS, tracked by ASYNCcnt (ISA 10./08_async).
// Low 32 bits of a generic pointer into __shared__ == LDS offset (aperture calc).
__device__ __forceinline__ void async_copy_b128(void* lds_dst, const void* gsrc) {
  uint32_t ldsaddr = (uint32_t)(uintptr_t)lds_dst;
  uint64_t gaddr   = (uint64_t)(uintptr_t)gsrc;
  asm volatile("global_load_async_to_lds_b128 %0, %1, off"
               :: "v"(ldsaddr), "v"(gaddr) : "memory");
}
__device__ __forceinline__ void async_wait0() {
  asm volatile("s_wait_asynccnt 0x0" ::: "memory");
}

// ---------------------------------------------------------------------------
// Generic bf16 WMMA GEMM:  D[M,N] = A[M,K] * W[N,K]^T
//   block tile 64(M) x 128(N), BK=32, 256 threads = 8 waves (4 M-tiles x 2)
//   each wave: one 16-row M tile x four 16-col N tiles
//   A staged through LDS:
//     - bf16 A: GLOBAL_LOAD_ASYNC_TO_LDS_B128 (ASYNCcnt path, no VGPR bounce)
//     - fp32 A: load + v_cvt_pk_bf16_f32 + ds_store (precision change in flight)
//   B streamed straight from L2 (weights are small and fully L2-resident).
// ---------------------------------------------------------------------------
template <bool A_IS_F32, bool D_IS_BF16>
__global__ __launch_bounds__(256) void wmma_gemm_kernel(
    const void* __restrict__ Ap, const bf16_t* __restrict__ Wp,
    void* __restrict__ Dp, int M, int N, int K) {
  const int tid  = threadIdx.x;
  const int lane = tid & 31;
  const int wid  = tid >> 5;
  const int wm   = wid & 3;        // wave's M tile (0..3)
  const int wn   = wid >> 2;       // wave's N half (0..1)
  const int m0   = blockIdx.y * 64;
  const int n0   = blockIdx.x * 128 + wn * 64;

  const int l15 = lane & 15;
  const int lhi = lane >> 4;       // 0 or 1

  __shared__ __align__(16) bf16_t shA[64 * 32];

  v8f acc[4] = {};

  const int arow = tid >> 2;        // 0..63
  const int acol = (tid & 3) * 8;   // 0,8,16,24

  for (int kk = 0; kk < K; kk += 32) {
    // ---- stage 64x32 A tile into LDS (16 B per thread) ----
    if (A_IS_F32) {
      const float* a = (const float*)Ap + (size_t)(m0 + arow) * K + kk + acol;
      f32x4 f0 = *(const f32x4*)a;
      f32x4 f1 = *(const f32x4*)(a + 4);
      bf16x8 h;
      h[0] = (bf16_t)f0[0]; h[1] = (bf16_t)f0[1];
      h[2] = (bf16_t)f0[2]; h[3] = (bf16_t)f0[3];
      h[4] = (bf16_t)f1[0]; h[5] = (bf16_t)f1[1];
      h[6] = (bf16_t)f1[2]; h[7] = (bf16_t)f1[3];
      *(bf16x8*)&shA[arow * 32 + acol] = h;
    } else {
      const bf16_t* a = (const bf16_t*)Ap + (size_t)(m0 + arow) * K + kk + acol;
      async_copy_b128(&shA[arow * 32 + acol], a);
      async_wait0();                 // ASYNCcnt==0 before the barrier
    }
    __syncthreads();

    // prefetch next K-slab of B into caches (global_prefetch_b8)
    if (kk + 32 < K)
      __builtin_prefetch(Wp + (size_t)(n0 + l15) * K + kk + 32, 0, 3);

    // ---- A fragment (ISA 7.12.2, 16-bit A 16x32): lane<16 -> K 0-7/16-23 ----
    const bf16_t* ap = &shA[(wm * 16 + l15) * 32 + lhi * 8];
    bf16x8 alo = *(const bf16x8*)ap;
    bf16x8 ahi = *(const bf16x8*)(ap + 16);
    v16bf afrag;
#pragma unroll
    for (int i = 0; i < 8; ++i) { afrag[i] = alo[i]; afrag[i + 8] = ahi[i]; }

    // ---- B fragments: lane = column, lane<16 -> K 0-7/16-23, lanes 16-31 -> K 8-15/24-31 ----
#pragma unroll
    for (int j = 0; j < 4; ++j) {
      const bf16_t* bp = Wp + (size_t)(n0 + j * 16 + l15) * K + kk + lhi * 8;
      bf16x8 blo = *(const bf16x8*)bp;
      bf16x8 bhi = *(const bf16x8*)(bp + 16);
      v16bf bfrag;
#pragma unroll
      for (int i = 0; i < 8; ++i) { bfrag[i] = blo[i]; bfrag[i + 8] = bhi[i]; }
      acc[j] = __builtin_amdgcn_wmma_f32_16x16x32_bf16(
          false, afrag, false, bfrag, (short)0, acc[j], false, false);
    }
    __syncthreads();
  }

  // ---- store D (C/D layout: VGPR v, lane<16 -> M=v, lane>=16 -> M=8+v) ----
#pragma unroll
  for (int j = 0; j < 4; ++j) {
    const int nn  = n0 + j * 16 + l15;
    const int mr0 = m0 + wm * 16 + lhi * 8;
#pragma unroll
    for (int v = 0; v < 8; ++v) {
      if (D_IS_BF16)
        ((bf16_t*)Dp)[(size_t)(mr0 + v) * N + nn] = (bf16_t)acc[j][v];
      else
        ((float*)Dp)[(size_t)(mr0 + v) * N + nn] = acc[j][v];
    }
  }
}

// ---------------------------------------------------------------------------
// Weight prep: W_comb = W_ih (+W_hh folded into first H cols) -> bf16,
//              bias = b_ih + b_hh, W_attT = W_att^T (fp32), W_embed -> bf16
// ---------------------------------------------------------------------------
__global__ __launch_bounds__(256) void prep_weights_kernel(
    const float* __restrict__ W_ih, const float* __restrict__ b_ih,
    const float* __restrict__ W_hh, const float* __restrict__ b_hh,
    const float* __restrict__ W_att, const float* __restrict__ W_embed,
    bf16_t* __restrict__ Wcomb, float* __restrict__ bias,
    float* __restrict__ WattT, bf16_t* __restrict__ Wembb) {
  const int idx = blockIdx.x * blockDim.x + threadIdx.x;
  if (idx < NG * KQ) {
    const int n = idx / KQ, k = idx % KQ;
    float w = W_ih[idx];
    if (k < H_) w += W_hh[n * H_ + k];
    Wcomb[idx] = (bf16_t)w;
  }
  if (idx < NG) bias[idx] = b_ih[idx] + b_hh[idx];
  if (idx < E_ * H_) {
    const int e = idx / H_, h = idx % H_;
    WattT[h * E_ + e] = W_att[idx];
    Wembb[idx] = (bf16_t)W_embed[idx];
  }
}

// ---------------------------------------------------------------------------
// State init: qs[b] = [q0 | r0 | x_0] (bf16), c = c0
// ---------------------------------------------------------------------------
__global__ __launch_bounds__(256) void init_state_kernel(
    const float* __restrict__ q0, const float* __restrict__ r0,
    const float* __restrict__ c0, const float* __restrict__ features,
    bf16_t* __restrict__ qs, float* __restrict__ c) {
  const int b = blockIdx.x, h = threadIdx.x;
  qs[(size_t)b * KQ + h]       = (bf16_t)q0[b * H_ + h];
  qs[(size_t)b * KQ + 256 + h] = (bf16_t)r0[b * E_ + h];
  qs[(size_t)b * KQ + 512 + h] = (bf16_t)features[(size_t)b * T_ * F_ + h];
  c[b * H_ + h] = c0[b * H_ + h];
}

// ---------------------------------------------------------------------------
// Per-step LSTM pointwise: gates -> (q,c), output projection, qp = q*W_att^T,
// stage x_{t+1} into q_star buffer. One block per molecule, 256 threads.
// ---------------------------------------------------------------------------
__global__ __launch_bounds__(256) void lstm_pointwise_kernel(
    const float* __restrict__ gates, const float* __restrict__ bias,
    float* __restrict__ c, bf16_t* __restrict__ qs, float* __restrict__ qp,
    float* __restrict__ out, const float* __restrict__ features,
    const int* __restrict__ lengths, const float* __restrict__ WattT,
    const float* __restrict__ Wfinal, const float* __restrict__ bfinal,
    int t) {
  const int b = blockIdx.x;
  const int h = threadIdx.x;
  __shared__ float shq[H_];
  __shared__ float red[256];

  const float* g = gates + (size_t)b * NG;
  const float ig = g[h]           + bias[h];
  const float fg = g[H_ + h]      + bias[H_ + h];
  const float gg = g[2 * H_ + h]  + bias[2 * H_ + h];
  const float og = g[3 * H_ + h]  + bias[3 * H_ + h];

  const float si = 1.f / (1.f + __expf(-ig));
  const float sf = 1.f / (1.f + __expf(-fg));
  const float so = 1.f / (1.f + __expf(-og));
  const float cc = sf * c[b * H_ + h] + si * tanhf(gg);
  const float q  = so * tanhf(cc);
  c[b * H_ + h] = cc;

  qs[(size_t)b * KQ + h] = (bf16_t)q;   // q for next-step gates GEMM
  shq[h] = q;
  if (t + 1 < T_)                       // stage x_{t+1}
    qs[(size_t)b * KQ + 512 + h] =
        (bf16_t)features[((size_t)b * T_ + t + 1) * F_ + h];

  // out[b,t] = relu(q) . W_final + b_final   (masked)
  red[h] = fmaxf(q, 0.f) * Wfinal[h];
  __syncthreads();
  for (int s = 128; s > 0; s >>= 1) {
    if (h < s) red[h] += red[h + s];
    __syncthreads();
  }
  if (h == 0)
    out[(size_t)b * T_ + t] = (t < lengths[b]) ? (red[0] + bfinal[0]) : 0.f;

  // qp[b,e] = sum_h q[h] * W_att[e,h]  (W_attT is [H,E] -> coalesced)
  float acc = 0.f;
#pragma unroll 4
  for (int k = 0; k < H_; ++k) acc += shq[k] * WattT[k * E_ + h];
  qp[(size_t)b * E_ + h] = acc;
}

// ---------------------------------------------------------------------------
// Per-step masked attention: e = Emb_b . qp_b, softmax, r = Emb_b^T . alpha
// One block per molecule, 256 threads = 8 waves; embedding is bf16/L2-resident.
// ---------------------------------------------------------------------------
__global__ __launch_bounds__(256) void attention_kernel(
    const bf16_t* __restrict__ emb, const float* __restrict__ qp,
    const int* __restrict__ lengths, bf16_t* __restrict__ qs) {
  const int b    = blockIdx.x;
  const int tid  = threadIdx.x;
  const int lane = tid & 31;
  const int wid  = tid >> 5;

  __shared__ float shqp[E_];
  __shared__ float she[T_];
  __shared__ float sha[T_];
  __shared__ float red[256];

  shqp[tid] = qp[(size_t)b * E_ + tid];
  __syncthreads();

  const int len = lengths[b];

  // scores: each wave handles 32 timesteps, lane l covers e = 8l..8l+7
  for (int j = 0; j < 32; ++j) {
    const int t = wid * 32 + j;
    const bf16_t* ep = emb + ((size_t)b * T_ + t) * E_ + lane * 8;
    bf16x8 ev = *(const bf16x8*)ep;
    float a = 0.f;
#pragma unroll
    for (int i = 0; i < 8; ++i) a += (float)ev[i] * shqp[lane * 8 + i];
#pragma unroll
    for (int off = 16; off > 0; off >>= 1) a += __shfl_xor(a, off, 32);
    if (lane == 0) she[t] = a;
  }
  __syncthreads();

  // masked softmax over T (len >= 1 guaranteed by the reference)
  red[tid] = (tid < len) ? she[tid] : -3.0e38f;
  __syncthreads();
  for (int s = 128; s > 0; s >>= 1) {
    if (tid < s) red[tid] = fmaxf(red[tid], red[tid + s]);
    __syncthreads();
  }
  const float m = red[0];
  __syncthreads();
  const float ex = (tid < len) ? __expf(she[tid] - m) : 0.f;
  red[tid] = ex;
  __syncthreads();
  for (int s = 128; s > 0; s >>= 1) {
    if (tid < s) red[tid] += red[tid + s];
    __syncthreads();
  }
  sha[tid] = ex / red[0];
  __syncthreads();

  // r[b,e] = sum_t alpha_t * emb[b,t,e]  (coalesced over e)
  float acc = 0.f;
  for (int t = 0; t < T_; ++t)
    acc += (float)emb[((size_t)b * T_ + t) * E_ + tid] * sha[t];
  qs[(size_t)b * KQ + 256 + tid] = (bf16_t)acc;  // r for next-step gates GEMM
}

// ---------------------------------------------------------------------------
extern "C" void kernel_launch(void* const* d_in, const int* in_sizes, int n_in,
                              void* d_out, int out_size, void* d_ws,
                              size_t ws_size, hipStream_t stream) {
  (void)in_sizes; (void)n_in; (void)out_size; (void)ws_size;

  const float* features = (const float*)d_in[0];
  const int*   lengths  = (const int*)d_in[1];
  const float* W_embed  = (const float*)d_in[2];
  const float* W_ih     = (const float*)d_in[3];
  const float* b_ih     = (const float*)d_in[4];
  const float* W_hh     = (const float*)d_in[5];
  const float* b_hh     = (const float*)d_in[6];
  const float* W_att    = (const float*)d_in[7];
  const float* W_final  = (const float*)d_in[8];
  const float* b_final  = (const float*)d_in[9];
  const float* q0       = (const float*)d_in[10];
  const float* c0       = (const float*)d_in[11];
  const float* r0       = (const float*)d_in[12];
  float* out = (float*)d_out;

  // workspace carve-out (~73 MB total)
  char* ws = (char*)d_ws;
  size_t off = 0;
  auto carve = [&](size_t bytes) -> void* {
    void* p = ws + off;
    off = (off + bytes + 255) & ~(size_t)255;
    return p;
  };
  bf16_t* emb   = (bf16_t*)carve((size_t)B_ * T_ * E_ * sizeof(bf16_t)); // 67 MB (L2-resident)
  bf16_t* Wcomb = (bf16_t*)carve((size_t)NG * KQ * sizeof(bf16_t));
  bf16_t* Wembb = (bf16_t*)carve((size_t)E_ * F_ * sizeof(bf16_t));
  float*  bias  = (float*)carve((size_t)NG * sizeof(float));
  float*  WattT = (float*)carve((size_t)H_ * E_ * sizeof(float));
  bf16_t* qs    = (bf16_t*)carve((size_t)B_ * KQ * sizeof(bf16_t));     // q_star [B,768]
  float*  gates = (float*)carve((size_t)B_ * NG * sizeof(float));
  float*  cbuf  = (float*)carve((size_t)B_ * H_ * sizeof(float));
  float*  qp    = (float*)carve((size_t)B_ * E_ * sizeof(float));

  // ---- setup ----
  prep_weights_kernel<<<(NG * KQ + 255) / 256, 256, 0, stream>>>(
      W_ih, b_ih, W_hh, b_hh, W_att, W_embed, Wcomb, bias, WattT, Wembb);
  init_state_kernel<<<B_, 256, 0, stream>>>(q0, r0, c0, features, qs, cbuf);

  // ---- embedding = features @ W_embed^T  (bf16 WMMA, D in bf16) ----
  {
    dim3 grid(E_ / 128, (B_ * T_) / 64);
    wmma_gemm_kernel<true, true><<<grid, 256, 0, stream>>>(
        (const void*)features, Wembb, (void*)emb, B_ * T_, E_, F_);
  }

  // ---- sequential scan over T steps ----
  for (int t = 0; t < T_; ++t) {
    dim3 ggrid(NG / 128, B_ / 64);  // gates = q_star @ W_comb^T
    wmma_gemm_kernel<false, false><<<ggrid, 256, 0, stream>>>(
        (const void*)qs, Wcomb, (void*)gates, B_, NG, KQ);
    lstm_pointwise_kernel<<<B_, 256, 0, stream>>>(
        gates, bias, cbuf, qs, qp, out, features, lengths, WattT, W_final,
        b_final, t);
    attention_kernel<<<B_, 256, 0, stream>>>(emb, qp, lengths, qs);
  }
}
